// WindowAttentionV2_28140625724103
// MI455X (gfx1250) — matrix operand order repaired and checked by reading, not compiled, not run on hardware
//
#include <hip/hip_runtime.h>
#include <hip/hip_bf16.h>
#include <math.h>

typedef __bf16 bf16_t;
typedef __attribute__((ext_vector_type(16))) __bf16 v16bf;
typedef __attribute__((ext_vector_type(8)))  float  v8f;

#define NTOK  64
#define DIMX  512
#define NHEAD 16
#define DHEAD 32
#define HIDN  384
#define NWIN  64

// LDS strides (bf16 elements) chosen for 16B alignment + bank-conflict-free frag reads
#define SX 520   // x / o tiles: 64 rows
#define SQ 40    // q / k tiles: 64 rows x 32
#define SV 72    // v^T tile: 32 rows x 64
#define SP 72    // p tile: 64 rows x 64 (aliases q+k region)

#define XBF_OFF  0
#define OBF_OFF  (NTOK * SX * 2)                    // 66560
#define WAVE_OFF (2 * NTOK * SX * 2)                // 133120
#define WAVE_SZ  (2 * NTOK * SQ * 2 + 32 * SV * 2)  // q + k + vT = 14848
#define SMEM_SZ  (WAVE_OFF + 8 * WAVE_SZ)           // 251904 bytes < 320KB

union FragU { v16bf v; uint4 q[2]; };

// A fragment (16x32 bf16, M x K): lane holds row m = mBase + (lane&15);
// elements 0..15 = K [16*khalf .. +15]
__device__ __forceinline__ v16bf load_a(const bf16_t* base, int stride,
                                        int mBase, int kBase, int ln, int kh) {
  const bf16_t* p = base + (mBase + ln) * stride + kBase + kh * 16;
  FragU u;
  u.q[0] = *(const uint4*)p;
  u.q[1] = *(const uint4*)(p + 8);
  return u.v;
}

// B fragment (32x16 bf16, K x N) from a row-major [n][k] source:
// lane holds col n = nBase + (lane&15), 16 contiguous K at kBase + 16*khalf
__device__ __forceinline__ v16bf load_b_rows(const bf16_t* base, int stride,
                                             int nBase, int kBase, int ln, int kh) {
  const bf16_t* p = base + (nBase + ln) * stride + kBase + kh * 16;
  FragU u;
  u.q[0] = *(const uint4*)p;
  u.q[1] = *(const uint4*)(p + 8);
  return u.v;
}

// B fragment from pre-packed global weights: [tile][lane][16 bf16]
__device__ __forceinline__ v16bf load_b_packed(const bf16_t* base, int tile, int lane) {
  const bf16_t* p = base + ((size_t)tile * 32 + lane) * 16;
  FragU u;
  u.q[0] = *(const uint4*)p;
  u.q[1] = *(const uint4*)(p + 8);
  return u.v;
}

// C fragment (8 f32, accumulator order) from pre-packed global: [..][lane][8]
__device__ __forceinline__ v8f load_c_packed(const float* p) {
  union { v8f v; float4 q[2]; } u;
  u.q[0] = *(const float4*)p;
  u.q[1] = *(const float4*)(p + 4);
  return u.v;
}

#define WMMA_BF16(A, B, C) \
  __builtin_amdgcn_wmma_f32_16x16x32_bf16(false, (A), false, (B), (short)0, (C), false, false)

// ---------------- setup kernels ----------------

// pack w_qkv [512][1536] f32 -> bf16 B-fragment order; tiles = kt*96 + nt
__global__ void pack_wqkv_k(const float* __restrict__ w, bf16_t* __restrict__ dst) {
  int gid = blockIdx.x * 256 + threadIdx.x;            // 786432
  int e = gid & 15, lane = (gid >> 4) & 31, tile = gid >> 9;
  int kt = tile / 96, nt = tile % 96;
  int n = nt * 16 + (lane & 15);
  int k = kt * 32 + (lane >> 4) * 16 + e;
  dst[gid] = (bf16_t)w[k * 1536 + n];
}

// pack w_out [512][512] f32 -> bf16; tiles = kt*32 + nt
__global__ void pack_wout_k(const float* __restrict__ w, bf16_t* __restrict__ dst) {
  int gid = blockIdx.x * 256 + threadIdx.x;            // 262144
  int e = gid & 15, lane = (gid >> 4) & 31, tile = gid >> 9;
  int kt = tile >> 5, nt = tile & 31;
  int n = nt * 16 + (lane & 15);
  int k = kt * 32 + (lane >> 4) * 16 + e;
  dst[gid] = (bf16_t)w[k * 512 + n];
}

// bias[h][i*64+j] = relu(rel_log @ w1 + b1) @ w2 + b2  (one thread per (i,j))
__global__ void bias_mlp_k(const float* __restrict__ rel, const float* __restrict__ w1,
                           const float* __restrict__ b1, const float* __restrict__ w2,
                           const float* __restrict__ b2, float* __restrict__ bias) {
  int p = blockIdx.x * 64 + threadIdx.x;               // 4096
  float x0 = rel[p * 2], x1 = rel[p * 2 + 1];
  float acc[16];
#pragma unroll
  for (int t = 0; t < 16; ++t) acc[t] = b2[t];
  for (int j = 0; j < HIDN; ++j) {
    float hv = fmaxf(x0 * w1[j] + x1 * w1[HIDN + j] + b1[j], 0.0f);
#pragma unroll
    for (int t = 0; t < 16; ++t) acc[t] += hv * w2[j * 16 + t];
  }
#pragma unroll
  for (int t = 0; t < 16; ++t) bias[t * 4096 + p] = acc[t];
}

// bm packed in WMMA C-fragment order:
// bm[w][h][mi][nj][lane][r] = bias[h][m][n] + mask[w][m][n],
//   m = mi*16 + (lane>>4)*8 + r, n = nj*16 + (lane&15)
__global__ void combine_bm_k(const float* __restrict__ bias, const float* __restrict__ mask,
                             float* __restrict__ bm) {
  int gid = blockIdx.x * 256 + threadIdx.x;            // 4194304
  int r = gid & 7, lane = (gid >> 3) & 31;
  int nj = (gid >> 8) & 3, mi = (gid >> 10) & 3;
  int h = (gid >> 12) & 15, w = gid >> 16;
  int m = mi * 16 + (lane >> 4) * 8 + r;
  int n = nj * 16 + (lane & 15);
  int e = m * 64 + n;
  bm[gid] = bias[h * 4096 + e] + mask[w * 4096 + e];
}

// ---------------- fused main kernel: one block = one window ----------------

__global__ __launch_bounds__(256, 1)
void wa_main(const float* __restrict__ x, const float* __restrict__ tau,
             const bf16_t* __restrict__ wqkv, const bf16_t* __restrict__ wout,
             const float* __restrict__ bmask, const float* __restrict__ bout,
             float* __restrict__ out) {
  extern __shared__ char smem[];
  bf16_t* xbf = (bf16_t*)(smem + XBF_OFF);
  bf16_t* obf = (bf16_t*)(smem + OBF_OFF);
  const int tid = threadIdx.x, wave = tid >> 5, lane = tid & 31;
  const int ln = lane & 15, kh = lane >> 4;
  bf16_t* qbf = (bf16_t*)(smem + WAVE_OFF + wave * WAVE_SZ);
  bf16_t* kbf = qbf + NTOK * SQ;
  bf16_t* vbf = kbf + NTOK * SQ;       // [32 dh][SV] transposed V
  bf16_t* pbf = qbf;                   // P aliases q+k (9216 <= 10240 B), used after S reads

  const int b = blockIdx.x;
  const int widx = b & (NWIN - 1);
  const float* xb = x + (size_t)b * NTOK * DIMX;

  // ---- stage x -> bf16 LDS (coalesced float4 loads) ----
  for (int i = tid; i < NTOK * DIMX / 4; i += 256) {
    float4 v = ((const float4*)xb)[i];
    int idx = i * 4, r = idx >> 9, c = idx & 511;
    bf16_t* d = xbf + r * SX + c;
    d[0] = (bf16_t)v.x; d[1] = (bf16_t)v.y; d[2] = (bf16_t)v.z; d[3] = (bf16_t)v.w;
  }
  __syncthreads();

  for (int hr = 0; hr < 2; ++hr) {
    const int h = wave * 2 + hr;
    const float tau_h = tau[h];

    // ---- qkv projection for this head: [64,512] @ [512,32] x3 ----
    for (int p = 0; p < 3; ++p) {
      const int ntBase = p * 32 + h * 2;
      const float scl = (p == 0) ? tau_h : 1.0f;
      for (int mi = 0; mi < 4; ++mi) {
        v8f a0 = {}, a1 = {};
        for (int kt = 0; kt < 16; ++kt) {
          v16bf a  = load_a(xbf, SX, mi * 16, kt * 32, ln, kh);
          v16bf b0 = load_b_packed(wqkv, kt * 96 + ntBase, lane);
          v16bf b1 = load_b_packed(wqkv, kt * 96 + ntBase + 1, lane);
          a0 = WMMA_BF16(a, b0, a0);
          a1 = WMMA_BF16(a, b1, a1);
        }
        if (p < 2) {  // q or k: L2-normalize rows over dh (16 lanes x 2 tiles), fold tau into q
          bf16_t* dst = (p == 0) ? qbf : kbf;
#pragma unroll
          for (int r = 0; r < 8; ++r) {
            float t = a0[r] * a0[r] + a1[r] * a1[r];
            t += __shfl_xor(t, 1, 32); t += __shfl_xor(t, 2, 32);
            t += __shfl_xor(t, 4, 32); t += __shfl_xor(t, 8, 32);
            float inv = scl / fmaxf(sqrtf(t), 1e-12f);
            int m = mi * 16 + kh * 8 + r;
            dst[m * SQ + ln]      = (bf16_t)(a0[r] * inv);
            dst[m * SQ + 16 + ln] = (bf16_t)(a1[r] * inv);
          }
        } else {      // v: store transposed [dh][token]
#pragma unroll
          for (int r = 0; r < 8; ++r) {
            int m = mi * 16 + kh * 8 + r;
            vbf[ln * SV + m]        = (bf16_t)a0[r];
            vbf[(16 + ln) * SV + m] = (bf16_t)a1[r];
          }
        }
      }
    }

    // ---- S = q k^T + (bias+mask) : bias enters as the WMMA C operand ----
    v8f s[4][4];
    const float* bmh = bmask + ((size_t)widx * NHEAD + h) * 4096;  // [mi][nj][lane][8]
#pragma unroll
    for (int mi = 0; mi < 4; ++mi) {
      v16bf a = load_a(qbf, SQ, mi * 16, 0, ln, kh);
#pragma unroll
      for (int nj = 0; nj < 4; ++nj) {
        v8f c = load_c_packed(bmh + ((mi * 4 + nj) * 32 + lane) * 8);
        v16bf bb = load_b_rows(kbf, SQ, nj * 16, 0, ln, kh);
        s[mi][nj] = WMMA_BF16(a, bb, c);
      }
    }
    // ---- softmax over j (4 tiles x 16 lanes per row), write P (bf16) ----
#pragma unroll
    for (int mi = 0; mi < 4; ++mi) {
#pragma unroll
      for (int r = 0; r < 8; ++r) {
        float mx = fmaxf(fmaxf(s[mi][0][r], s[mi][1][r]), fmaxf(s[mi][2][r], s[mi][3][r]));
        mx = fmaxf(mx, __shfl_xor(mx, 1, 32)); mx = fmaxf(mx, __shfl_xor(mx, 2, 32));
        mx = fmaxf(mx, __shfl_xor(mx, 4, 32)); mx = fmaxf(mx, __shfl_xor(mx, 8, 32));
        float e0 = __expf(s[mi][0][r] - mx), e1 = __expf(s[mi][1][r] - mx);
        float e2 = __expf(s[mi][2][r] - mx), e3 = __expf(s[mi][3][r] - mx);
        float sum = e0 + e1 + e2 + e3;
        sum += __shfl_xor(sum, 1, 32); sum += __shfl_xor(sum, 2, 32);
        sum += __shfl_xor(sum, 4, 32); sum += __shfl_xor(sum, 8, 32);
        float inv = 1.0f / sum;
        int m = mi * 16 + kh * 8 + r;
        pbf[m * SP + ln]      = (bf16_t)(e0 * inv);
        pbf[m * SP + 16 + ln] = (bf16_t)(e1 * inv);
        pbf[m * SP + 32 + ln] = (bf16_t)(e2 * inv);
        pbf[m * SP + 48 + ln] = (bf16_t)(e3 * inv);
      }
    }

    // ---- O_h = P @ V : [64,64] @ [64,32] ----
    for (int mi = 0; mi < 4; ++mi) {
      v8f a0 = {}, a1 = {};
#pragma unroll
      for (int kt = 0; kt < 2; ++kt) {
        v16bf a  = load_a(pbf, SP, mi * 16, kt * 32, ln, kh);
        v16bf b0 = load_b_rows(vbf, SV, 0,  kt * 32, ln, kh);
        v16bf b1 = load_b_rows(vbf, SV, 16, kt * 32, ln, kh);
        a0 = WMMA_BF16(a, b0, a0);
        a1 = WMMA_BF16(a, b1, a1);
      }
#pragma unroll
      for (int r = 0; r < 8; ++r) {
        int m = mi * 16 + kh * 8 + r;
        obf[m * SX + h * 32 + ln]      = (bf16_t)a0[r];
        obf[m * SX + h * 32 + 16 + ln] = (bf16_t)a1[r];
      }
    }
  }

  __syncthreads();

  // ---- out = O[64,512] @ w_out[512,512] + b_out; wave owns 64 columns ----
  for (int mi = 0; mi < 4; ++mi) {
    v8f acc[4] = {{}, {}, {}, {}};
    for (int kt = 0; kt < 16; ++kt) {
      v16bf a = load_a(obf, SX, mi * 16, kt * 32, ln, kh);
#pragma unroll
      for (int t = 0; t < 4; ++t) {
        v16bf bb = load_b_packed(wout, kt * 32 + wave * 4 + t, lane);
        acc[t] = WMMA_BF16(a, bb, acc[t]);
      }
    }
#pragma unroll
    for (int t = 0; t < 4; ++t) {
      int col = (wave * 4 + t) * 16 + ln;
      float bo = bout[col];
#pragma unroll
      for (int r = 0; r < 8; ++r) {
        int m = mi * 16 + kh * 8 + r;
        out[(size_t)b * NTOK * DIMX + m * DIMX + col] = acc[t][r] + bo;
      }
    }
  }
}

// ---------------- host launcher ----------------

extern "C" void kernel_launch(void* const* d_in, const int* in_sizes, int n_in,
                              void* d_out, int out_size, void* d_ws, size_t ws_size,
                              hipStream_t stream) {
  (void)in_sizes; (void)n_in; (void)out_size; (void)ws_size;
  const float* x    = (const float*)d_in[0];
  const float* mask = (const float*)d_in[1];
  const float* wqkv = (const float*)d_in[2];
  const float* tau  = (const float*)d_in[3];
  const float* mw1  = (const float*)d_in[4];
  const float* mb1  = (const float*)d_in[5];
  const float* mw2  = (const float*)d_in[6];
  const float* mb2  = (const float*)d_in[7];
  const float* wout = (const float*)d_in[8];
  const float* bo   = (const float*)d_in[9];
  const float* rel  = (const float*)d_in[10];
  float* out = (float*)d_out;

  char* ws = (char*)d_ws;
  bf16_t* wqkv_p = (bf16_t*)ws;                 // 1,572,864 B
  bf16_t* wout_p = (bf16_t*)(ws + 1572864);     //   524,288 B
  float*  bias_t = (float*)(ws + 2097152);      //   262,144 B
  float*  bm     = (float*)(ws + 2359296);      // 16,777,216 B  (total ~18.3 MB)

  (void)hipFuncSetAttribute((const void*)wa_main,
                            hipFuncAttributeMaxDynamicSharedMemorySize, SMEM_SZ);

  pack_wqkv_k<<<786432 / 256, 256, 0, stream>>>(wqkv, wqkv_p);
  pack_wout_k<<<262144 / 256, 256, 0, stream>>>(wout, wout_p);
  bias_mlp_k<<<64, 64, 0, stream>>>(rel, mw1, mb1, mw2, mb2, bias_t);
  combine_bm_k<<<4194304 / 256, 256, 0, stream>>>(bias_t, mask, bm);
  wa_main<<<2048, 256, SMEM_SZ, stream>>>(x, tau, wqkv_p, wout_p, bm, bo, out);
}
